// SequenceLSTM_62929860821333
// MI455X (gfx1250) — matrix operand order repaired and checked by reading, not compiled, not run on hardware
//
#include <hip/hip_runtime.h>
#include <hip/hip_bf16.h>

#define H    51
#define HP   64
#define NG   256           // 4 * HP padded gate width
#define BT   16            // batch rows per workgroup
#define TSEQ 2048
#define FUT  64
#define TTOT (TSEQ + FUT)
#define A1S  72            // halfs per row (144B, multiple of 16B)
#define A2S  136           // halfs per row (272B, multiple of 16B)
#define GS   260           // f32 per gate-buffer row (conflict-skewed)

typedef __attribute__((ext_vector_type(16))) _Float16 v16h;
typedef __attribute__((ext_vector_type(8)))  _Float16 v8h;
typedef __attribute__((ext_vector_type(8)))  float    v8f;

// ---------------- fast transcendentals ------------------------------------
__device__ __forceinline__ float fast_exp(float x) {
#if __has_builtin(__builtin_amdgcn_exp2f)
  return __builtin_amdgcn_exp2f(x * 1.4426950408889634f);
#else
  return __expf(x);
#endif
}
__device__ __forceinline__ float fast_sig(float x) {
#if __has_builtin(__builtin_amdgcn_rcpf)
  return __builtin_amdgcn_rcpf(1.0f + fast_exp(-x));
#else
  return 1.0f / (1.0f + fast_exp(-x));
#endif
}
__device__ __forceinline__ float fast_tanh(float x) {
#if __has_builtin(__builtin_amdgcn_tanhf)
  return __builtin_amdgcn_tanhf(x);     // hardware v_tanh_f32 (confirmed in asm)
#else
  return 2.0f * fast_sig(2.0f * x) - 1.0f;
#endif
}

// ---------------- weight repack: f32 -> f16 [n][k], input+bias folded ------
// WT1: 256 x 64   (k: 0..50 = Whh1, 51 = Wih1, 52 = bias)
// WT2: 256 x 128  (k: 0..50 = Wih2, 51..101 = Whh2, 102 = bias)
__global__ void lstm_pack_w(const float* __restrict__ Wih1, const float* __restrict__ Whh1,
                            const float* __restrict__ bih1, const float* __restrict__ bhh1,
                            const float* __restrict__ Wih2, const float* __restrict__ Whh2,
                            const float* __restrict__ bih2, const float* __restrict__ bhh2,
                            _Float16* __restrict__ WT1, _Float16* __restrict__ WT2) {
  int idx = blockIdx.x * 256 + threadIdx.x;
  if (idx < NG * 64) {                       // layer-1 table
    int n = idx >> 6, k = idx & 63;
    int g = n >> 6, c = n & 63;
    float v = 0.0f;
    if (c < H) {
      int gc = g * H + c;
      if (k < H)        v = Whh1[gc * H + k];
      else if (k == H)  v = Wih1[gc];
      else if (k == 52) v = bih1[gc] + bhh1[gc];
    }
    WT1[idx] = (_Float16)v;
  }
  int idx2 = idx - NG * 64;
  if (idx2 >= 0 && idx2 < NG * 128) {        // layer-2 table
    int n = idx2 >> 7, k = idx2 & 127;
    int g = n >> 6, c = n & 63;
    float v = 0.0f;
    if (c < H) {
      int gc = g * H + c;
      if (k < H)          v = Wih2[gc * H + k];
      else if (k < 2 * H) v = Whh2[gc * H + (k - H)];
      else if (k == 102)  v = bih2[gc] + bhh2[gc];
    }
    WT2[idx2] = (_Float16)v;
  }
}

// A-fragment load: lane holds row M=lane&15; K runs {0-15} or {16-31}
__device__ __forceinline__ v16h load_a(const _Float16* rowp, int ks, int hi16) {
  const int r0 = ks * 32 + hi16 * 16;
  v16h a;
  *reinterpret_cast<v8h*>(&a)       = *reinterpret_cast<const v8h*>(rowp + r0);
  *(reinterpret_cast<v8h*>(&a) + 1) = *reinterpret_cast<const v8h*>(rowp + r0 + 8);
  return a;
}

#define WMMA_F16(acc, a, b) \
  acc = __builtin_amdgcn_wmma_f32_16x16x32_f16(false, a, false, b, (short)0, acc, false, false)

// Branchless activation + scatter to f32 gate buffer.
//   sigmoid(x) = fma(tanh(0.5x), 0.5, 0.5)   -> pre=0.5, mul=0.5, add=0.5
//   tanh(x)    = fma(tanh(x),    1.0, 0.0)   -> pre=1.0, mul=1.0, add=0.0
__device__ __forceinline__ void store_gates(float* Gbuf, v8f acc, int nt, int lane,
                                            float pre, float mul, float add) {
  const int col   = nt * 16 + (lane & 15);
  const int mbase = (lane >> 4) * 8;
#pragma unroll
  for (int r = 0; r < 8; ++r) {
    float y = fast_tanh(pre * acc[r]);
    Gbuf[(mbase + r) * GS + col] = __builtin_fmaf(y, mul, add);
  }
}

// ---------------- persistent scan kernel: 16 blocks x 256 threads ----------
__global__ __launch_bounds__(256, 1)
void lstm_scan(const float* __restrict__ x,
               const _Float16* __restrict__ WT1, const _Float16* __restrict__ WT2,
               const float* __restrict__ fcw, const float* __restrict__ fcb,
               float* __restrict__ out) {
  __shared__ __align__(16) _Float16 A1[BT * A1S];   // [h1 | x | 1]   K=64
  __shared__ __align__(16) _Float16 A2[BT * A2S];   // [h1 | h2 | 1]  K=128
  __shared__ float G1[BT * GS];
  __shared__ float G2[BT * GS];
  __shared__ float C1[BT * HP];
  __shared__ float C2[BT * HP];
  __shared__ float H2s[BT * HP];
  __shared__ float XIN[BT];
  __shared__ float OFEED[BT];
  __shared__ float FCW[HP];
  __shared__ float FCB;

  const int tid  = threadIdx.x;
  const int lane = tid & 31;
  const int wv   = tid >> 5;         // 8 waves
  const int lr   = lane & 15;        // matrix row / column within tile
  const int hi16 = lane >> 4;
  const int b0   = blockIdx.x * BT;
  const int m_e  = tid >> 4;         // elementwise row
  const int s_e  = tid & 15;

  // Per-wave, loop-invariant activation constants (gate = wv>>1: i,f,g,o)
  const bool isg   = ((wv >> 1) == 2);
  const float pre  = isg ? 1.0f : 0.5f;
  const float amul = isg ? 1.0f : 0.5f;
  const float aadd = isg ? 0.0f : 0.5f;

  // --- zero state ---
  for (int i = tid; i < BT * A1S; i += 256) A1[i] = (_Float16)0.0f;
  for (int i = tid; i < BT * A2S; i += 256) A2[i] = (_Float16)0.0f;
  for (int i = tid; i < BT * HP; i += 256) { C1[i] = 0.0f; C2[i] = 0.0f; H2s[i] = 0.0f; }
  __syncthreads();
  if (wv == 0) {
    if (lane < BT) {
      A1[lane * A1S + 52]  = (_Float16)1.0f;     // bias lane
      A2[lane * A2S + 102] = (_Float16)1.0f;
      XIN[lane] = x[(b0 + lane) * TSEQ];         // first input
    }
  }
  if (tid < HP) FCW[tid] = (tid < H) ? fcw[tid] : 0.0f;
  if (tid == 0) FCB = fcb[0];

  // --- pin this wave's weight B-fragments in registers (lane=col n,
  //     16 contiguous k halfs) ---
  v16h B1[2][2], B2[2][4];
#pragma unroll
  for (int ti = 0; ti < 2; ++ti) {
    const int n = (wv * 2 + ti) * 16 + lr;
#pragma unroll
    for (int ks = 0; ks < 2; ++ks) {
      const _Float16* p = WT1 + n * 64 + ks * 32 + hi16 * 16;
      v16h v;
      *reinterpret_cast<v8h*>(&v)       = *reinterpret_cast<const v8h*>(p);
      *(reinterpret_cast<v8h*>(&v) + 1) = *reinterpret_cast<const v8h*>(p + 8);
      B1[ti][ks] = v;
    }
#pragma unroll
    for (int ks = 0; ks < 4; ++ks) {
      const _Float16* p = WT2 + n * 128 + ks * 32 + hi16 * 16;
      v16h v;
      *reinterpret_cast<v8h*>(&v)       = *reinterpret_cast<const v8h*>(p);
      *(reinterpret_cast<v8h*>(&v) + 1) = *reinterpret_cast<const v8h*>(p + 8);
      B2[ti][ks] = v;
    }
  }
  __syncthreads();

  for (int t = 0; t < TTOT; ++t) {
    // prefetch next timestep's input (hidden behind this step's compute)
    float xnext = 0.0f;
    if (wv == 0) {
      if (lane < BT && (t + 1) < TSEQ) xnext = x[(b0 + lane) * TSEQ + (t + 1)];
      if (lane < BT) A1[lane * A1S + 51] = (_Float16)XIN[lane];
    }
    __syncthreads();

    // ---- layer 1 GEMM: [16x64] x [64x256] ----
    {
      const _Float16* arow = &A1[lr * A1S];
      v16h a0 = load_a(arow, 0, hi16);
      v16h a1 = load_a(arow, 1, hi16);
      v8f acc0 = {}; v8f acc1 = {};
      WMMA_F16(acc0, a0, B1[0][0]); WMMA_F16(acc0, a1, B1[0][1]);
      WMMA_F16(acc1, a0, B1[1][0]); WMMA_F16(acc1, a1, B1[1][1]);
      store_gates(G1, acc0, wv * 2 + 0, lane, pre, amul, aadd);
      store_gates(G1, acc1, wv * 2 + 1, lane, pre, amul, aadd);
    }
    __syncthreads();

    // ---- layer 1 cell update ----
    for (int j = s_e; j < H; j += 16) {
      float iv = G1[m_e * GS + j];
      float fv = G1[m_e * GS + 64 + j];
      float gv = G1[m_e * GS + 128 + j];
      float ov = G1[m_e * GS + 192 + j];
      float c  = __builtin_fmaf(fv, C1[m_e * HP + j], iv * gv);
      C1[m_e * HP + j] = c;
      _Float16 h = (_Float16)(ov * fast_tanh(c));
      A1[m_e * A1S + j] = h;
      A2[m_e * A2S + j] = h;
    }
    __syncthreads();

    // ---- layer 2 GEMM: [16x128] x [128x256] ----
    {
      const _Float16* arow = &A2[lr * A2S];
      v8f acc0 = {}; v8f acc1 = {};
#pragma unroll
      for (int ks = 0; ks < 4; ++ks) {
        v16h a = load_a(arow, ks, hi16);
        WMMA_F16(acc0, a, B2[0][ks]);
        WMMA_F16(acc1, a, B2[1][ks]);
      }
      store_gates(G2, acc0, wv * 2 + 0, lane, pre, amul, aadd);
      store_gates(G2, acc1, wv * 2 + 1, lane, pre, amul, aadd);
    }
    __syncthreads();

    // ---- layer 2 cell update ----
    for (int j = s_e; j < H; j += 16) {
      float iv = G2[m_e * GS + j];
      float fv = G2[m_e * GS + 64 + j];
      float gv = G2[m_e * GS + 128 + j];
      float ov = G2[m_e * GS + 192 + j];
      float c  = __builtin_fmaf(fv, C2[m_e * HP + j], iv * gv);
      C2[m_e * HP + j] = c;
      float h = ov * fast_tanh(c);
      A2[m_e * A2S + H + j] = (_Float16)h;
      H2s[m_e * HP + j] = h;
    }
    __syncthreads();

    // ---- fc head: out = h2 . fcw + fcb  (16-lane shfl reduction) ----
    float part = 0.0f;
    for (int j = s_e; j < H; j += 16) part += H2s[m_e * HP + j] * FCW[j];
    part += __shfl_xor(part, 8);
    part += __shfl_xor(part, 4);
    part += __shfl_xor(part, 2);
    part += __shfl_xor(part, 1);
    if (s_e == 0) {
      float ov = part + FCB;
      out[(b0 + m_e) * TTOT + t] = ov;
      OFEED[m_e] = ov;
    }
    __syncthreads();

    if (wv == 0) {
      if (lane < BT) XIN[lane] = ((t + 1) < TSEQ) ? xnext : OFEED[lane];
    }
  }
}

extern "C" void kernel_launch(void* const* d_in, const int* in_sizes, int n_in,
                              void* d_out, int out_size, void* d_ws, size_t ws_size,
                              hipStream_t stream) {
  const float* x    = (const float*)d_in[0];
  const float* Wih1 = (const float*)d_in[1];
  const float* Whh1 = (const float*)d_in[2];
  const float* bih1 = (const float*)d_in[3];
  const float* bhh1 = (const float*)d_in[4];
  const float* Wih2 = (const float*)d_in[5];
  const float* Whh2 = (const float*)d_in[6];
  const float* bih2 = (const float*)d_in[7];
  const float* bhh2 = (const float*)d_in[8];
  const float* fcw  = (const float*)d_in[9];
  const float* fcb  = (const float*)d_in[10];
  (void)in_sizes; (void)n_in; (void)out_size; (void)ws_size;

  _Float16* WT1 = (_Float16*)d_ws;           // 256*64  f16 = 32 KB
  _Float16* WT2 = WT1 + NG * 64;             // 256*128 f16 = 64 KB

  lstm_pack_w<<<(NG * 64 + NG * 128 + 255) / 256, 256, 0, stream>>>(
      Wih1, Whh1, bih1, bhh1, Wih2, Whh2, bih2, bhh2, WT1, WT2);

  lstm_scan<<<256 / BT, 256, 0, stream>>>(x, WT1, WT2, fcw, fcb, (float*)d_out);
}
